// ChannelMerger_47339129536903
// MI455X (gfx1250) — matrix operand order repaired and checked by reading, not compiled, not run on hardware
//
#include <hip/hip_runtime.h>
#include <hip/hip_bf16.h>
#include <math.h>

typedef _Float16 h8  __attribute__((ext_vector_type(8)));
typedef _Float16 h16 __attribute__((ext_vector_type(16)));
typedef float    f8  __attribute__((ext_vector_type(8)));

#define B_  128
#define C_  273
#define T_  3072
#define S_  4
#define O_  270
#define D_  2048
#define KP  288      // C padded to multiple of 32
#define MP  272      // O padded to multiple of 16
#define NT  128      // t-tile width per block
#define LDT 296      // LDS row stride in halves (pad for bank spread)
#define RPAD 1088    // (S_*O_=1080) rounded up to 16*68
#define NQ  4        // D-chunks for partial-score reduction

#define EMB_BYTES    (D_*KP*4)          // embT [2048][288] fp32 = 2,359,296 B
#define W_ELEMS      (S_*MP*KP)         // 313,344 halves per array
#define W_BYTES      (W_ELEMS*2)

#if __has_builtin(__builtin_amdgcn_sched_group_barrier)
#define SGB(mask, size, id) __builtin_amdgcn_sched_group_barrier((mask), (size), (id))
#else
#define SGB(mask, size, id)
#endif
#define MASK_VMEM_RD 0x020
#define MASK_WMMA    (0x008 | 0x400)    // MFMA/WMMA | TRANS (gfx1250 classes WMMA as XDL/TRANS)

// ---------------- Kernel A1: Fourier embedding (transposed) ----------------
__global__ void emb_kernel(const float* __restrict__ positions, float* __restrict__ embT) {
    const int c = blockIdx.x;                      // channel 0..272
    const float px = positions[2*c+0] + 0.2f;
    const float py = positions[2*c+1] + 0.2f;
    const float step = 6.28318530717958647692f / 1.4f;  // 2*pi/width
    for (int k = threadIdx.x; k < 1024; k += blockDim.x) {
        const int i = k >> 5, j = k & 31;
        const float loc = px * (step * (float)i) + py * (step * (float)j);
        float sv, cv;
        sincosf(loc, &sv, &cv);
        embT[(size_t)k        * KP + c] = cv;      // emb[c, k]        = cos
        embT[(size_t)(1024+k) * KP + c] = sv;      // emb[c, 1024+k]   = sin
    }
}

// ------- Kernel A2a: partial scores over a 512-wide D chunk -------
__global__ __launch_bounds__(288)
void scores_part_kernel(const float* __restrict__ embT, const float* __restrict__ heads,
                        float* __restrict__ scoresP) {
    const int c     = threadIdx.x;        // 0..287
    const int rbase = blockIdx.x * 16;    // 16 (s,o) rows per block
    const int q     = blockIdx.y;         // D chunk 0..3

    int rows[16];
    #pragma unroll
    for (int g = 0; g < 16; ++g) {
        int r = rbase + g;
        rows[g] = (r < S_*O_) ? r : (S_*O_ - 1);   // clamp to stay in-bounds
    }

    float acc[16];
    #pragma unroll
    for (int g = 0; g < 16; ++g) acc[g] = 0.f;

    const int d0 = q * (D_ / NQ);
    #pragma unroll 4
    for (int dd = 0; dd < D_ / NQ; ++dd) {
        const int d = d0 + dd;
        const float e = embT[(size_t)d * KP + c];          // coalesced
        #pragma unroll
        for (int g = 0; g < 16; ++g)
            acc[g] += e * heads[(size_t)rows[g] * D_ + d]; // uniform -> scalar loads
    }

    #pragma unroll
    for (int g = 0; g < 16; ++g) {
        const int r = rbase + g;
        scoresP[((size_t)q * RPAD + r) * KP + c] = acc[g];
    }
}

// ------- Kernel A2b: sum partials, softmax over c, f16 hi/lo split -------
__global__ __launch_bounds__(288)
void softmax_split_kernel(const float* __restrict__ scoresP,
                          _Float16* __restrict__ Whi, _Float16* __restrict__ Wlo) {
    __shared__ float sm[16];
    const int c  = threadIdx.x;          // 0..287
    const bool cv = (c < C_);
    const int rbase = blockIdx.x * 16;
    const int lane = c & 31, wv = c >> 5;   // 9 waves of 32

    for (int g = 0; g < 16; ++g) {
        const int r = rbase + g;
        float sc = 0.f;
        #pragma unroll
        for (int q = 0; q < NQ; ++q)
            sc += scoresP[((size_t)q * RPAD + r) * KP + c];
        if (!cv) sc = -3.0e38f;

        // block max
        float m = sc;
        for (int off = 16; off > 0; off >>= 1) m = fmaxf(m, __shfl_down(m, off, 32));
        if (lane == 0) sm[wv] = m;
        __syncthreads();
        if (wv == 0) {
            float x = (lane < 9) ? sm[lane] : -3.0e38f;
            for (int off = 8; off > 0; off >>= 1) x = fmaxf(x, __shfl_down(x, off, 32));
            if (lane == 0) sm[0] = x;
        }
        __syncthreads();
        const float mx = sm[0];
        __syncthreads();

        // block sum of exp
        const float e = cv ? expf(sc - mx) : 0.f;
        float s = e;
        for (int off = 16; off > 0; off >>= 1) s += __shfl_down(s, off, 32);
        if (lane == 0) sm[wv] = s;
        __syncthreads();
        if (wv == 0) {
            float x = (lane < 9) ? sm[lane] : 0.f;
            for (int off = 8; off > 0; off >>= 1) x += __shfl_down(x, off, 32);
            if (lane == 0) sm[0] = x;
        }
        __syncthreads();
        const float tot = sm[0];
        __syncthreads();

        if (r < S_*O_) {
            const float w = e / tot;
            const _Float16 hi = (_Float16)w;
            const _Float16 lo = (_Float16)(w - (float)hi);
            const int si = r / O_, oi = r % O_;
            const size_t off_ = ((size_t)(si * MP + oi)) * KP + c;
            Whi[off_] = cv ? hi : (_Float16)0.f;   // c in [273,288) padded with zeros
            Wlo[off_] = cv ? lo : (_Float16)0.f;
        }
    }
}

// ---------------- Kernel A3: zero the o = 270,271 pad rows ----------------
__global__ void padw_kernel(_Float16* __restrict__ Whi, _Float16* __restrict__ Wlo) {
    const int i = blockIdx.x * blockDim.x + threadIdx.x;
    const int total = S_ * 2 * KP;
    if (i >= total) return;
    const int s   = i / (2 * KP);
    const int rem = i % (2 * KP);
    const int o   = O_ + rem / KP;
    const int c   = rem % KP;
    const size_t off = ((size_t)(s * MP + o)) * KP + c;
    Whi[off] = (_Float16)0.f;
    Wlo[off] = (_Float16)0.f;
}

// -------- Kernel B: out[b,o,t] = sum_c W[s_b,o,c] * meg[b,c,t]  (f16x2-split WMMA) --------
__global__ __launch_bounds__(256)
void mix_gemm_kernel(const float* __restrict__ meg, const int* __restrict__ subject,
                     const _Float16* __restrict__ Whi, const _Float16* __restrict__ Wlo,
                     float* __restrict__ out) {
    __shared__ _Float16 sH[NT * LDT];   // meg tile, hi halves, [t][c] layout
    __shared__ _Float16 sL[NT * LDT];   // meg tile, lo halves

    const int b   = blockIdx.y;
    const int t0  = blockIdx.x * NT;
    const int tid = threadIdx.x;
    const int s   = subject[b];

    // ---- Stage meg tile -> LDS with fp32 -> (f16 hi, f16 lo) split ----
    {
        const int col = tid & (NT - 1);   // t offset within tile
        const int rh  = tid >> 7;         // 0..1 (two rows per pass)
        #pragma unroll 4
        for (int it = 0; it < KP / 2; ++it) {
            const int row = 2 * it + rh;  // c index 0..287
            float x = 0.f;
            if (row < C_)
                x = __builtin_nontemporal_load(&meg[((size_t)b * C_ + row) * T_ + t0 + col]);
            const _Float16 hi = (_Float16)x;
            const _Float16 lo = (_Float16)(x - (float)hi);
            sH[col * LDT + row] = hi;
            sL[col * LDT + row] = lo;
        }
    }
    __syncthreads();

    const int lane = tid & 31;
    const int n0   = (tid >> 5) * 16;     // each of 8 waves owns a 16-wide N slice
    const int lm   = lane & 15;
    const int lh   = lane >> 4;           // 0/1 half-wave

    const _Float16* bh  = &sH[(n0 + lm) * LDT];
    const _Float16* bl  = &sL[(n0 + lm) * LDT];
    const _Float16* ah0 = Whi + (size_t)(s * MP + lm) * KP;
    const _Float16* al0 = Wlo + (size_t)(s * MP + lm) * KP;

    union HU { h16 v; h8 p[2]; };

    for (int m = 0; m < MP / 16; ++m) {               // 17 M-chunks of 16 rows
        const _Float16* ah = ah0 + (size_t)m * 16 * KP;
        const _Float16* al = al0 + (size_t)m * 16 * KP;
        f8 acc0 = {0.f, 0.f, 0.f, 0.f, 0.f, 0.f, 0.f, 0.f};
        f8 acc1 = {0.f, 0.f, 0.f, 0.f, 0.f, 0.f, 0.f, 0.f};

        // double-buffered A fragments: prefetch k+1 while computing k
        HU A0h, A0l, A1h, A1l;
        {
            const int ka = lh * 8;
            A0h.p[0] = *(const h8*)(ah + ka);
            A0h.p[1] = *(const h8*)(ah + ka + 16);
            A0l.p[0] = *(const h8*)(al + ka);
            A0l.p[1] = *(const h8*)(al + ka + 16);
        }
        SGB(MASK_VMEM_RD, 4, 0);                      // prologue loads first
        #pragma unroll
        for (int k = 0; k < KP / 32; ++k) {           // 9 K-steps of 32
            HU& cH = (k & 1) ? A1h : A0h;
            HU& cL = (k & 1) ? A1l : A0l;
            HU& nH = (k & 1) ? A0h : A1h;
            HU& nL = (k & 1) ? A0l : A1l;
            if (k < KP / 32 - 1) {
                const int ka = (k + 1) * 32 + lh * 8;
                nH.p[0] = *(const h8*)(ah + ka);
                nH.p[1] = *(const h8*)(ah + ka + 16);
                nL.p[0] = *(const h8*)(al + ka);
                nL.p[1] = *(const h8*)(al + ka + 16);
            }
            // B (32x16 f16): lanes 0-15 hold K k..k+7,k+16..k+23; lanes 16-31 hold K k+8..k+15,k+24..k+31
            const int kb = k * 32 + lh * 8;
            HU Bh, Bl;
            Bh.p[0] = *(const h8*)(bh + kb);
            Bh.p[1] = *(const h8*)(bh + kb + 16);
            Bl.p[0] = *(const h8*)(bl + kb);
            Bl.p[1] = *(const h8*)(bl + kb + 16);
            // D += Ah*Bh + Ah*Bl + Al*Bh  (fp32 accumulate; lo*lo term negligible)
            // two alternating accumulator chains to relax WMMA->WMMA dependencies
            if (k & 1) {
                acc1 = __builtin_amdgcn_wmma_f32_16x16x32_f16(false, cH.v, false, Bh.v,
                                                              (short)0, acc1, false, false);
                acc0 = __builtin_amdgcn_wmma_f32_16x16x32_f16(false, cH.v, false, Bl.v,
                                                              (short)0, acc0, false, false);
                acc1 = __builtin_amdgcn_wmma_f32_16x16x32_f16(false, cL.v, false, Bh.v,
                                                              (short)0, acc1, false, false);
            } else {
                acc0 = __builtin_amdgcn_wmma_f32_16x16x32_f16(false, cH.v, false, Bh.v,
                                                              (short)0, acc0, false, false);
                acc1 = __builtin_amdgcn_wmma_f32_16x16x32_f16(false, cH.v, false, Bl.v,
                                                              (short)0, acc1, false, false);
                acc0 = __builtin_amdgcn_wmma_f32_16x16x32_f16(false, cL.v, false, Bh.v,
                                                              (short)0, acc0, false, false);
            }
            // pipeline spec: issue next-step A loads ahead of this step's WMMAs
            if (k < KP / 32 - 1) SGB(MASK_VMEM_RD, 4, 0);
            SGB(MASK_WMMA, 3, 0);
        }
        const f8 acc = acc0 + acc1;
        // D layout: VGPR r holds row (o0 + r + 8*half-wave), lane&15 = column
        const int o0 = m * 16;
        #pragma unroll
        for (int r = 0; r < 8; ++r) {
            const int row = o0 + r + lh * 8;
            if (row < O_)
                __builtin_nontemporal_store(acc[r],
                    &out[((size_t)b * O_ + row) * T_ + t0 + n0 + lm]);
        }
    }
}

// ------------------------------- launcher -------------------------------
extern "C" void kernel_launch(void* const* d_in, const int* in_sizes, int n_in,
                              void* d_out, int out_size, void* d_ws, size_t ws_size,
                              hipStream_t stream) {
    const float* meg       = (const float*)d_in[0];
    const float* positions = (const float*)d_in[1];
    const float* heads     = (const float*)d_in[2];
    const int*   subject   = (const int*)d_in[3];
    float*       out       = (float*)d_out;

    char* ws = (char*)d_ws;
    float*    embT    = (float*)ws;                               // [2048][288] fp32
    _Float16* Whi     = (_Float16*)(ws + EMB_BYTES);              // [4][272][288] f16
    _Float16* Wlo     = Whi + W_ELEMS;                            // [4][272][288] f16
    float*    scoresP = (float*)(ws + EMB_BYTES + 2*W_BYTES);     // [4][1088][288] fp32

    emb_kernel<<<C_, 256, 0, stream>>>(positions, embT);

    dim3 gscore(RPAD / 16, NQ);
    scores_part_kernel<<<gscore, 288, 0, stream>>>(embT, heads, scoresP);
    softmax_split_kernel<<<RPAD / 16, 288, 0, stream>>>(scoresP, Whi, Wlo);
    padw_kernel<<<(S_*2*KP + 255) / 256, 256, 0, stream>>>(Whi, Wlo);

    dim3 grid(T_ / NT, B_);
    mix_gemm_kernel<<<grid, 256, 0, stream>>>(meg, subject, Whi, Wlo, out);
}
